// SelfAttention_90452011253823
// MI455X (gfx1250) — matrix operand order repaired and checked by reading, not compiled, not run on hardware
//
#include <hip/hip_runtime.h>
#include <hip/hip_bf16.h>

#define BATCH 8
#define SEQ   1024
#define EMB   1024
#define NH    16
#define HD    64

typedef __attribute__((ext_vector_type(16))) __bf16 bf16x16;
typedef __attribute__((ext_vector_type(8)))  float  f32x8;
typedef unsigned int u32x4 __attribute__((ext_vector_type(4)));
typedef int          i32x4 __attribute__((ext_vector_type(4)));
typedef int          i32x8 __attribute__((ext_vector_type(8)));

#if defined(__HIP_DEVICE_COMPILE__) && __has_builtin(__builtin_amdgcn_tensor_load_to_lds)
#define HAVE_TDM 1
#else
#define HAVE_TDM 0
#endif

static __device__ inline f32x8 zero8() {
  f32x8 z;
#pragma unroll
  for (int i = 0; i < 8; ++i) z[i] = 0.f;
  return z;
}

// A-matrix 16x32 bf16 fragment (contiguous k order used here, matching frag_b): lane holds row (lane&15);
// elements 0..7 = K (lane>>4)*16 .. +7 ; elements 8..15 = K (lane>>4)*16+8 .. +7
static __device__ inline bf16x16 frag_a(const __bf16* p0, const __bf16* p1) {
  bf16x16 a;
#pragma unroll
  for (int j = 0; j < 8; ++j) { a[j] = p0[j]; a[8 + j] = p1[j]; }
  return a;
}

// B-matrix 32x16 bf16 fragment: lane holds column (lane&15);
// elements 0..15 = K (lane>>4)*16 .. +15  (contiguous)
static __device__ inline bf16x16 frag_b(const __bf16* p) {
  bf16x16 b;
#pragma unroll
  for (int j = 0; j < 16; ++j) b[j] = p[j];
  return b;
}

#define WMMA_BF16(a, b, c) \
  __builtin_amdgcn_wmma_f32_16x16x32_bf16(false, (a), false, (b), (short)0, (c), false, false)

#if HAVE_TDM
// TDM 2D tile load: global (row-major, elem stride `row_stride`) -> LDS,
// tile = tile_k elems x tile_rows rows, 2-byte elements.
// D# group0/group1 bitfields per CDNA5 ISA §8.3/8.4.
static __device__ inline void tdm_load_2d(unsigned int lds_off, const void* gaddr,
                                          unsigned int tile_k, unsigned int tile_rows,
                                          unsigned int row_stride) {
  unsigned long long ga = (unsigned long long)(uintptr_t)gaddr;
  u32x4 g0;
  g0[0] = 1u;                                             // count=1 (valid user D#)
  g0[1] = lds_off;                                        // lds_addr (bytes)
  g0[2] = (unsigned int)(ga & 0xffffffffu);               // global_addr[31:0]
  g0[3] = (unsigned int)((ga >> 32) & 0x01ffffffu) | (2u << 30);  // addr[56:32], type=2
  i32x8 g1;
  g1[0] = (int)(1u << 16);                                // data_size=1 (2 bytes)
  g1[1] = (int)((tile_k & 0xffffu) << 16);                // tensor_dim0[15:0]
  g1[2] = (int)(((tile_k >> 16) & 0xffffu) | ((tile_rows & 0xffffu) << 16));
  g1[3] = (int)(((tile_rows >> 16) & 0xffffu) | ((tile_k & 0xffffu) << 16));  // tile_dim0
  g1[4] = (int)(tile_rows & 0xffffu);                     // tile_dim1 (tile_dim2=0)
  g1[5] = (int)row_stride;                                // tensor_dim0_stride[31:0]
  g1[6] = 0;
  g1[7] = 0;
  i32x4 z4;
  z4[0] = 0; z4[1] = 0; z4[2] = 0; z4[3] = 0;
#if __clang_major__ >= 23
  i32x8 z8;
#pragma unroll
  for (int i = 0; i < 8; ++i) z8[i] = 0;
  __builtin_amdgcn_tensor_load_to_lds(g0, g1, z4, z4, z8, 0);
#else
  __builtin_amdgcn_tensor_load_to_lds(g0, g1, z4, z4, 0);
#endif
}
#endif

// Shared bf16-WMMA GEMM mainloop: block tile 128x128, 8 waves, wave 64x32.
// A tile staged by TDM (wave 0 issues descriptor, TENSORcnt-gated); B tile
// staged transposed ([n][k]) by all threads.
static __device__ inline void gemm_mainloop(const __bf16* __restrict__ A,
                                            const __bf16* __restrict__ Bm, int ldb,
                                            int m0, int n0, __bf16* As, __bf16* Bs,
                                            f32x8 acc[4][2], int tid, int wid,
                                            int r, int hi2, int wm, int wn) {
  for (int k0 = 0; k0 < EMB; k0 += 32) {
    __syncthreads();  // previous iteration's fragment reads complete
#if HAVE_TDM
    if (wid == 0) {
      tdm_load_2d((unsigned int)(uintptr_t)As,
                  &A[(size_t)m0 * EMB + k0], 32u, 128u, (unsigned int)EMB);
    }
#else
#pragma unroll
    for (int c = 0; c < 2; ++c) {  // A tile 128x32, 16B per thread-chunk
      int id = tid + 256 * c;
      int row = id >> 2, ko = (id & 3) * 8;
      *(uint4*)&As[row * 32 + ko] =
          *(const uint4*)&A[(size_t)(m0 + row) * EMB + k0 + ko];
    }
#endif
#pragma unroll
    for (int c = 0; c < 4; ++c) {  // B tile 32x128, stored transposed [n][k]
      int id = tid + 256 * c;
      int kk = id >> 5, n4 = (id & 31) * 4;
      ushort4 u = *(const ushort4*)&Bm[(size_t)(k0 + kk) * ldb + n0 + n4];
      unsigned short* bs = (unsigned short*)Bs;
      bs[(n4 + 0) * 32 + kk] = u.x;
      bs[(n4 + 1) * 32 + kk] = u.y;
      bs[(n4 + 2) * 32 + kk] = u.z;
      bs[(n4 + 3) * 32 + kk] = u.w;
    }
    if (k0 + 32 < EMB)  // speculative prefetch of next B tile into GL2
      __builtin_prefetch(&Bm[(size_t)(k0 + 32) * ldb + n0 + (tid & 31) * 4], 0, 1);
#if HAVE_TDM
    if (wid == 0) __builtin_amdgcn_s_wait_tensorcnt(0);
#endif
    __syncthreads();
    bf16x16 af[4], bfr[2];
#pragma unroll
    for (int tm = 0; tm < 4; ++tm) {
      int row = wm * 64 + tm * 16 + r;
      af[tm] = frag_a(&As[row * 32 + hi2 * 16], &As[row * 32 + hi2 * 16 + 8]);
    }
#pragma unroll
    for (int tn = 0; tn < 2; ++tn) {
      int cc = wn * 32 + tn * 16 + r;
      bfr[tn] = frag_b(&Bs[cc * 32 + hi2 * 16]);
    }
#pragma unroll
    for (int tm = 0; tm < 4; ++tm)
#pragma unroll
      for (int tn = 0; tn < 2; ++tn)
        acc[tm][tn] = WMMA_BF16(af[tm], bfr[tn], acc[tm][tn]);
  }
}

// ---------------------------------------------------------------- f32 -> bf16
__global__ __launch_bounds__(256) void cvt_kernel(const float* __restrict__ in,
                                                  __bf16* __restrict__ out, int n4) {
  int i = blockIdx.x * 256 + threadIdx.x;
  if (i < n4) {
    float4 v = ((const float4*)in)[i];
    out[4 * i + 0] = (__bf16)v.x;
    out[4 * i + 1] = (__bf16)v.y;
    out[4 * i + 2] = (__bf16)v.z;
    out[4 * i + 3] = (__bf16)v.w;
  }
}

// ------------------------------------------------- QKV GEMM (8192x3072x1024)
__global__ __launch_bounds__(256) void qkv_gemm_kernel(
    const __bf16* __restrict__ A, const __bf16* __restrict__ Bm,
    const float* __restrict__ bias,
    __bf16* __restrict__ qb, __bf16* __restrict__ kb, __bf16* __restrict__ vb) {
  __shared__ __attribute__((aligned(16))) __bf16 As[128 * 32];
  __shared__ __attribute__((aligned(16))) __bf16 Bs[128 * 32];
  const int tid = threadIdx.x;
  const int lane = tid & 31, wid = tid >> 5;
  const int r = lane & 15, hi2 = lane >> 4;
  const int wm = wid >> 2, wn = wid & 3;
  const int m0 = blockIdx.x * 128, n0 = blockIdx.y * 128;

  f32x8 acc[4][2];
#pragma unroll
  for (int tm = 0; tm < 4; ++tm)
#pragma unroll
    for (int tn = 0; tn < 2; ++tn) acc[tm][tn] = zero8();

  gemm_mainloop(A, Bm, 3 * EMB, m0, n0, As, Bs, acc, tid, wid, r, hi2, wm, wn);

  // scatter epilogue: qkv column -> (which, head, dd); row -> (batch, l)
#pragma unroll
  for (int tn = 0; tn < 2; ++tn) {
    int col = n0 + wn * 32 + tn * 16 + r;
    int which = col >> 10, c = col & 1023, h = c >> 6, dd = c & 63;
    __bf16* dst = (which == 0) ? qb : ((which == 1) ? kb : vb);
    float bv = bias[col];
#pragma unroll
    for (int tm = 0; tm < 4; ++tm) {
#pragma unroll
      for (int p = 0; p < 8; ++p) {
        int row = m0 + wm * 64 + tm * 16 + hi2 * 8 + p;
        int bb = row >> 10, l = row & 1023;
        dst[(((size_t)bb * NH + h) * SEQ + l) * HD + dd] =
            (__bf16)(acc[tm][tn][p] + bv);
      }
    }
  }
}

// --------------------------------- pass 1: per-COLUMN (query-axis) max & sum
// grid (B*H, L/128); each wave owns 16 columns, streams rows i >= column base
__global__ __launch_bounds__(256) void attn_stats_kernel(
    const __bf16* __restrict__ qg, const __bf16* __restrict__ kg,
    float* __restrict__ cmax, float* __restrict__ csum) {
  const int bh = blockIdx.x;
  const int tid = threadIdx.x, lane = tid & 31, wid = tid >> 5;
  const int r = lane & 15, hi2 = lane >> 4;
  const int jw = blockIdx.y * 128 + wid * 16;
  const int jcol = jw + r;
  const __bf16* qp = qg + (size_t)bh * SEQ * HD;
  const __bf16* kp = kg + (size_t)bh * SEQ * HD;
  const float scale = 0.125f;  // 1/sqrt(64)

  const __bf16* krow = &kp[(size_t)(jw + r) * HD];
  bf16x16 bk0 = frag_b(krow + hi2 * 16);       // K(d) 0..31
  bf16x16 bk1 = frag_b(krow + 32 + hi2 * 16);  // K(d) 32..63

  float m = -INFINITY, Z = 0.f;
  for (int i0 = jw; i0 < SEQ; i0 += 16) {
    const __bf16* qrow = &qp[(size_t)(i0 + r) * HD];
    bf16x16 a0 = frag_a(qrow + hi2 * 16, qrow + hi2 * 16 + 8);
    bf16x16 a1 = frag_a(qrow + 32 + hi2 * 16, qrow + 40 + hi2 * 16);
    f32x8 c = zero8();
    c = WMMA_BF16(a0, bk0, c);
    c = WMMA_BF16(a1, bk1, c);
    float tmax = -INFINITY, vals[8];
#pragma unroll
    for (int p = 0; p < 8; ++p) {
      int i = i0 + hi2 * 8 + p;
      float s = (i >= jcol) ? c[p] * scale : -INFINITY;  // causal mask
      vals[p] = s;
      tmax = fmaxf(tmax, s);
    }
    float nm = fmaxf(m, tmax);
    if (nm != -INFINITY) {  // online rescale
      float za = 0.f;
#pragma unroll
      for (int p = 0; p < 8; ++p) za += __expf(vals[p] - nm);
      Z = Z * __expf(m - nm) + za;
      m = nm;
    }
  }
  // merge the two row-halves (lane r and lane r+16 hold the same column)
  float mo = __shfl_xor(m, 16, 32);
  float Zo = __shfl_xor(Z, 16, 32);
  float nm = fmaxf(m, mo);
  float Zt = Z * __expf(m - nm) + Zo * __expf(mo - nm);
  if (lane < 16) {
    cmax[(size_t)bh * SEQ + jcol] = nm;
    csum[(size_t)bh * SEQ + jcol] = Zt;
  }
}

// ------------------------- pass 2: y = column-normalized(P) @ V, per (b,h)
__global__ __launch_bounds__(256) void attn_av_kernel(
    const __bf16* __restrict__ qg, const __bf16* __restrict__ kg,
    const __bf16* __restrict__ vg, const float* __restrict__ cmax,
    const float* __restrict__ csum, __bf16* __restrict__ yb) {
  __shared__ __attribute__((aligned(16))) __bf16 Pl[8][16 * 32];  // per-wave P tile
  const int bh = blockIdx.x;
  const int b = bh >> 4, h = bh & 15;
  const int tid = threadIdx.x, lane = tid & 31, wid = tid >> 5;
  const int r = lane & 15, hi2 = lane >> 4;
  const int iw = blockIdx.y * 128 + wid * 16;  // wave's 16 query rows
  const __bf16* qp = qg + (size_t)bh * SEQ * HD;
  const __bf16* kp = kg + (size_t)bh * SEQ * HD;
  const __bf16* vp = vg + (size_t)bh * SEQ * HD;
  const float scale = 0.125f;

  const __bf16* qrow = &qp[(size_t)(iw + r) * HD];
  bf16x16 aq0 = frag_a(qrow + hi2 * 16, qrow + hi2 * 16 + 8);
  bf16x16 aq1 = frag_a(qrow + 32 + hi2 * 16, qrow + 40 + hi2 * 16);

  f32x8 acc[4];
#pragma unroll
  for (int t = 0; t < 4; ++t) acc[t] = zero8();

  __bf16* Pw = Pl[wid];

  for (int j0 = 0; j0 < iw + 16; j0 += 32) {
#pragma unroll
    for (int js = 0; js < 2; ++js) {
      int jb = j0 + js * 16;
      if (jb < iw + 16) {  // uniform per wave
        const __bf16* krow = &kp[(size_t)(jb + r) * HD];
        bf16x16 bk0 = frag_b(krow + hi2 * 16);
        bf16x16 bk1 = frag_b(krow + 32 + hi2 * 16);
        f32x8 s = zero8();
        s = WMMA_BF16(aq0, bk0, s);
        s = WMMA_BF16(aq1, bk1, s);
        int jc = jb + r;
        float mj = cmax[(size_t)bh * SEQ + jc];
        float rz = 1.f / csum[(size_t)bh * SEQ + jc];
#pragma unroll
        for (int p = 0; p < 8; ++p) {
          int i = iw + hi2 * 8 + p;
          float pv = (i >= jc) ? __expf(s[p] * scale - mj) * rz : 0.f;
          Pw[(hi2 * 8 + p) * 32 + js * 16 + r] = (__bf16)pv;  // transpose via LDS
        }
      } else {
#pragma unroll
        for (int p = 0; p < 8; ++p)
          Pw[(hi2 * 8 + p) * 32 + js * 16 + r] = (__bf16)0.f;
      }
    }
    // P as A fragment (16x32), V as B fragments (32 x 16 per dd-tile)
    bf16x16 ap = frag_a(&Pw[r * 32 + hi2 * 16], &Pw[r * 32 + hi2 * 16 + 8]);
#pragma unroll
    for (int t4 = 0; t4 < 4; ++t4) {
      bf16x16 bv;
#pragma unroll
      for (int t = 0; t < 16; ++t) {
        int vr = j0 + hi2 * 16 + t;
        vr = vr < SEQ ? vr : SEQ - 1;  // clamp; P==0 nullifies masked tail
        bv[t] = vp[(size_t)vr * HD + t4 * 16 + r];
      }
      acc[t4] = WMMA_BF16(ap, bv, acc[t4]);
    }
  }
#pragma unroll
  for (int t4 = 0; t4 < 4; ++t4) {
#pragma unroll
    for (int p = 0; p < 8; ++p) {
      int l = iw + hi2 * 8 + p;
      int c = h * HD + t4 * 16 + r;
      yb[((size_t)b * SEQ + l) * EMB + c] = (__bf16)acc[t4][p];
    }
  }
}

// ------------------------------------------------ proj GEMM (8192x1024x1024)
__global__ __launch_bounds__(256) void proj_gemm_kernel(
    const __bf16* __restrict__ A, const __bf16* __restrict__ Bm,
    const float* __restrict__ bias, float* __restrict__ out) {
  __shared__ __attribute__((aligned(16))) __bf16 As[128 * 32];
  __shared__ __attribute__((aligned(16))) __bf16 Bs[128 * 32];
  const int tid = threadIdx.x;
  const int lane = tid & 31, wid = tid >> 5;
  const int r = lane & 15, hi2 = lane >> 4;
  const int wm = wid >> 2, wn = wid & 3;
  const int m0 = blockIdx.x * 128, n0 = blockIdx.y * 128;

  f32x8 acc[4][2];
#pragma unroll
  for (int tm = 0; tm < 4; ++tm)
#pragma unroll
    for (int tn = 0; tn < 2; ++tn) acc[tm][tn] = zero8();

  gemm_mainloop(A, Bm, EMB, m0, n0, As, Bs, acc, tid, wid, r, hi2, wm, wn);

#pragma unroll
  for (int tn = 0; tn < 2; ++tn) {
    int col = n0 + wn * 32 + tn * 16 + r;
    float bv = bias[col];
#pragma unroll
    for (int tm = 0; tm < 4; ++tm) {
#pragma unroll
      for (int p = 0; p < 8; ++p) {
        int row = m0 + wm * 64 + tm * 16 + hi2 * 8 + p;
        out[(size_t)row * EMB + col] = acc[tm][tn][p] + bv;
      }
    }
  }
}

extern "C" void kernel_launch(void* const* d_in, const int* in_sizes, int n_in,
                              void* d_out, int out_size, void* d_ws, size_t ws_size,
                              hipStream_t stream) {
  (void)in_sizes; (void)n_in; (void)out_size; (void)ws_size;
  const float* x     = (const float*)d_in[0];
  const float* Wqkv  = (const float*)d_in[1];
  const float* bqkv  = (const float*)d_in[2];
  const float* Wproj = (const float*)d_in[3];
  const float* bproj = (const float*)d_in[4];
  float* out = (float*)d_out;

  char* ws = (char*)d_ws;
  size_t off = 0;
  auto alloc = [&](size_t bytes) -> void* {
    void* p = ws + off;
    off += (bytes + 255) & ~(size_t)255;
    return p;
  };
  __bf16* xb     = (__bf16*)alloc((size_t)BATCH * SEQ * EMB * 2);
  __bf16* wqkvb  = (__bf16*)alloc((size_t)EMB * 3 * EMB * 2);
  __bf16* wprojb = (__bf16*)alloc((size_t)EMB * EMB * 2);
  __bf16* qb     = (__bf16*)alloc((size_t)BATCH * NH * SEQ * HD * 2);
  __bf16* kb     = (__bf16*)alloc((size_t)BATCH * NH * SEQ * HD * 2);
  __bf16* vb     = (__bf16*)alloc((size_t)BATCH * NH * SEQ * HD * 2);
  __bf16* yb     = (__bf16*)alloc((size_t)BATCH * SEQ * EMB * 2);
  float*  cm     = (float*)alloc((size_t)BATCH * NH * SEQ * 4);
  float*  cs     = (float*)alloc((size_t)BATCH * NH * SEQ * 4);

  int n;
  n = BATCH * SEQ * EMB / 4;
  cvt_kernel<<<(n + 255) / 256, 256, 0, stream>>>(x, xb, n);
  n = EMB * 3 * EMB / 4;
  cvt_kernel<<<(n + 255) / 256, 256, 0, stream>>>(Wqkv, wqkvb, n);
  n = EMB * EMB / 4;
  cvt_kernel<<<(n + 255) / 256, 256, 0, stream>>>(Wproj, wprojb, n);

  qkv_gemm_kernel<<<dim3(BATCH * SEQ / 128, 3 * EMB / 128), 256, 0, stream>>>(
      xb, wqkvb, bqkv, qb, kb, vb);
  attn_stats_kernel<<<dim3(BATCH * NH, SEQ / 128), 256, 0, stream>>>(qb, kb, cm, cs);
  attn_av_kernel<<<dim3(BATCH * NH, SEQ / 128), 256, 0, stream>>>(qb, kb, vb, cm, cs, yb);
  proj_gemm_kernel<<<dim3(BATCH * SEQ / 128, EMB / 128), 256, 0, stream>>>(
      yb, wprojb, bproj, out);
}
